// Pointer_65343632441663
// MI455X (gfx1250) — matrix operand order repaired and checked by reading, not compiled, not run on hardware
//
#include <hip/hip_runtime.h>

// Problem sizes (fixed by the reference).
#define Bb 16
#define Ss 4096
#define Dd 1024
#define TILE_M 64          // s-rows staged per workgroup
#define ROWB 2064          // LDS bytes per staged row: 1024 bf16 + 8 elem pad (bank-conflict free)

typedef __bf16 bf16_t;
typedef bf16_t v8bf  __attribute__((ext_vector_type(8)));
typedef bf16_t v16bf __attribute__((ext_vector_type(16)));
typedef float  v8f   __attribute__((ext_vector_type(8)));

__device__ __forceinline__ unsigned pack2bf(float a, float b) {
    union { bf16_t h[2]; unsigned u; } x;
    x.h[0] = (bf16_t)a; x.h[1] = (bf16_t)b;
    return x.u;
}

// Branch-free tanh. Prefer hardware V_TANH_F32 (TRANS op, co-executes with WMMA);
// fallback: tanh(x) = 1 - 2/(exp2(x*2*log2e)+1)  -> v_exp_f32 + v_rcp_f32, no branches,
// saturates correctly to +/-1 at +/-inf.
__device__ __forceinline__ float fast_tanh(float x) {
#if __has_builtin(__builtin_amdgcn_tanhf)
    return __builtin_amdgcn_tanhf(x);
#else
    float e2 = __builtin_amdgcn_exp2f(x * 2.8853900817779268f); // 2*log2(e)
    float r  = __builtin_amdgcn_rcpf(e2 + 1.0f);
    return __builtin_fmaf(-2.0f, r, 1.0f);
#endif
}

// WMMA 16-bit fragment loaders (layouts per CDNA5 ISA 7.12.2):
//   A (16x32): row M = lane&15; per-lane K chunks at 8*ksel and 16+8*ksel
//   B (32x16): col N = lane&15; per-lane K chunks at 8*ksel and 16+8*ksel
__device__ __forceinline__ v16bf load_a_frag(const char* aRow, int k0, int ksel) {
    int ka = k0 + 8 * ksel;
    v8bf lo = *(const v8bf*)(aRow + ka * 2);
    v8bf hi = *(const v8bf*)(aRow + (ka + 16) * 2);
    return __builtin_shufflevector(lo, hi, 0,1,2,3,4,5,6,7,8,9,10,11,12,13,14,15);
}
__device__ __forceinline__ v16bf load_b_frag(const bf16_t* wrow, int k0, int ksel) {
    int kb = k0 + 8 * ksel;
    v8bf lo = *(const v8bf*)(wrow + kb);
    v8bf hi = *(const v8bf*)(wrow + kb + 16);
    return __builtin_shufflevector(lo, hi, 0,1,2,3,4,5,6,7,8,9,10,11,12,13,14,15);
}
__device__ __forceinline__ v8f wmma_bf16(v16bf a, v16bf b, v8f c) {
    return __builtin_amdgcn_wmma_f32_16x16x32_bf16(false, a, false, b, (short)0, c,
                                                   false, false);
}

// ---------------------------------------------------------------------------
// Prep 1: Wk (f32, [D,D]) -> bf16 copy in workspace. 2 MB, stays L2-resident.
// ---------------------------------------------------------------------------
__global__ __launch_bounds__(256) void cvt_wk_kernel(const float* __restrict__ wk,
                                                     bf16_t* __restrict__ wkb) {
    int idx = blockIdx.x * 256 + threadIdx.x;        // one float4 each
    float4 x = ((const float4*)wk)[idx];
    uint2 p;
    p.x = pack2bf(x.x, x.y);
    p.y = pack2bf(x.z, x.w);
    *(uint2*)(wkb + (size_t)idx * 4) = p;
}

// ---------------------------------------------------------------------------
// Prep 2: q[b,e] = dot(target[b,:], Wq[e,:]).  16384 dots of length 1024.
// ---------------------------------------------------------------------------
__global__ __launch_bounds__(256) void q_kernel(const float* __restrict__ target,
                                                const float* __restrict__ wq,
                                                float* __restrict__ qbuf) {
    int idx = blockIdx.x * 256 + threadIdx.x;        // 0 .. 16383
    int b = idx >> 10;
    int e = idx & (Dd - 1);
    const float4* t = (const float4*)(target + (size_t)b * Dd);
    const float4* w = (const float4*)(wq + (size_t)e * Dd);
    float s = 0.f;
#pragma unroll 8
    for (int i = 0; i < Dd / 4; ++i) {
        float4 a = t[i], c = w[i];
        s = fmaf(a.x, c.x, s); s = fmaf(a.y, c.y, s);
        s = fmaf(a.z, c.z, s); s = fmaf(a.w, c.w, s);
    }
    qbuf[idx] = s;
}

// ---------------------------------------------------------------------------
// Main fused kernel: scores[b,s] = sum_e v[e] * tanh(q[b,e] + (memory@Wk^T)[b,s,e])
// Grid: (S/TILE_M, B), 256 threads = 8 waves.
//   wave w: subtile = w&3 (16 s-rows), ehalf = w>>2 (half of the 64 N-tiles).
// Each wave processes 2 e-tiles per pass sharing one A-fragment stream:
//   per 2 WMMAs: 2 ds_load_b128 (A) + 4 global_load_b128 (B), software-pipelined.
// ---------------------------------------------------------------------------
__global__ __launch_bounds__(256) void bahdanau_main(const float* __restrict__ memory,
                                                     const bf16_t* __restrict__ wkb,
                                                     const float* __restrict__ qbuf,
                                                     const float* __restrict__ vvec,
                                                     float* __restrict__ scores) {
    extern __shared__ char smem[];                    // TILE_M*ROWB bytes A-stage + 64 f32 combine
    float* combine = (float*)(smem + TILE_M * ROWB);

    const int b    = blockIdx.y;
    const int s0   = blockIdx.x * TILE_M;
    const int tid  = threadIdx.x;
    const int lane = tid & 31;
    const int wave = tid >> 5;
    const int subtile = wave & 3;
    const int ehalf   = wave >> 2;

    if (tid < TILE_M) combine[tid] = 0.0f;

    // ---- Stage A: 64 rows x 1024 f32 -> bf16 in LDS (padded rows) ----
    const float4* src = (const float4*)(memory + ((size_t)b * Ss + s0) * Dd);
#pragma unroll 4
    for (int i = 0; i < 64; ++i) {
        int f = tid + 256 * i;                        // float4 index, 0..16383
        float4 x = src[f];
        int e4  = f << 2;
        int row = e4 >> 10;
        int col = e4 & (Dd - 1);
        uint2 p;
        p.x = pack2bf(x.x, x.y);
        p.y = pack2bf(x.z, x.w);
        *(uint2*)(smem + row * ROWB + col * 2) = p;
    }
    __syncthreads();

    const int colq = lane & 15;
    const int ksel = lane >> 4;
    const char* aRow = smem + (size_t)(subtile * 16 + colq) * ROWB;

    float acc[8];
#pragma unroll
    for (int i = 0; i < 8; ++i) acc[i] = 0.0f;

    for (int et = 0; et < 32; et += 2) {
        const int ecol0 = (ehalf * 32 + et) * 16 + colq;
        const int ecol1 = ecol0 + 16;
        const float qv0 = qbuf[b * Dd + ecol0];
        const float vv0 = vvec[ecol0];
        const float qv1 = qbuf[b * Dd + ecol1];
        const float vv1 = vvec[ecol1];
        const bf16_t* wrow0 = wkb + (size_t)ecol0 * Dd;
        const bf16_t* wrow1 = wkb + (size_t)ecol1 * Dd;

        v8f c0 = {0.f,0.f,0.f,0.f,0.f,0.f,0.f,0.f};
        v8f c1 = {0.f,0.f,0.f,0.f,0.f,0.f,0.f,0.f};

        // Software-pipelined K loop: fragments for step k+32 are issued before the
        // WMMAs consuming step k, so s_wait overlaps with matrix math.
        v16bf a  = load_a_frag(aRow, 0, ksel);
        v16bf b0 = load_b_frag(wrow0, 0, ksel);
        v16bf b1 = load_b_frag(wrow1, 0, ksel);
#pragma unroll 4
        for (int k0 = 32; k0 < Dd; k0 += 32) {
            v16bf na  = load_a_frag(aRow, k0, ksel);
            v16bf nb0 = load_b_frag(wrow0, k0, ksel);
            v16bf nb1 = load_b_frag(wrow1, k0, ksel);
            c0 = wmma_bf16(a, b0, c0);
            c1 = wmma_bf16(a, b1, c1);
            a = na; b0 = nb0; b1 = nb1;
        }
        c0 = wmma_bf16(a, b0, c0);
        c1 = wmma_bf16(a, b1, c1);

        // Fused epilogue: k never hits memory. Branch-free tanh (TRANS co-exec).
#pragma unroll
        for (int i = 0; i < 8; ++i) {
            acc[i] = fmaf(vv0, fast_tanh(qv0 + c0[i]), acc[i]);
            acc[i] = fmaf(vv1, fast_tanh(qv1 + c1[i]), acc[i]);
        }
    }

    // Reduce each row-partial across the 16 lanes of each half-wave.
#pragma unroll
    for (int i = 0; i < 8; ++i) {
        float v = acc[i];
        v += __shfl_xor(v, 8);
        v += __shfl_xor(v, 4);
        v += __shfl_xor(v, 2);
        v += __shfl_xor(v, 1);
        acc[i] = v;
    }
    if ((lane & 15) == 0) {
        int rbase = subtile * 16 + ksel * 8;
#pragma unroll
        for (int i = 0; i < 8; ++i)
            atomicAdd(&combine[rbase + i], acc[i]);   // ds f32 atomic; merges the two e-halves
    }
    __syncthreads();
    if (tid < TILE_M)
        scores[(size_t)b * Ss + s0 + tid] = combine[tid];
}

// ---------------------------------------------------------------------------
// Masked softmax over S per batch. mask is JAX bool -> 1 byte per element.
// ---------------------------------------------------------------------------
__global__ __launch_bounds__(256) void softmax_kernel(const float* __restrict__ scores,
                                                      const unsigned char* __restrict__ mask,
                                                      float* __restrict__ out) {
    __shared__ float red[256];
    const int b = blockIdx.x;
    const int tid = threadIdx.x;
    float vals[Ss / 256];
    float mx = -3.0e38f;
#pragma unroll
    for (int i = 0; i < Ss / 256; ++i) {
        int s = tid + 256 * i;
        float v = scores[(size_t)b * Ss + s] - 1.0e9f * (float)mask[(size_t)b * Ss + s];
        vals[i] = v;
        mx = fmaxf(mx, v);
    }
    red[tid] = mx; __syncthreads();
    for (int off = 128; off > 0; off >>= 1) {
        if (tid < off) red[tid] = fmaxf(red[tid], red[tid + off]);
        __syncthreads();
    }
    mx = red[0]; __syncthreads();

    float sum = 0.f;
#pragma unroll
    for (int i = 0; i < Ss / 256; ++i) {
        vals[i] = __expf(vals[i] - mx);
        sum += vals[i];
    }
    red[tid] = sum; __syncthreads();
    for (int off = 128; off > 0; off >>= 1) {
        if (tid < off) red[tid] += red[tid + off];
        __syncthreads();
    }
    float inv = 1.0f / red[0];
#pragma unroll
    for (int i = 0; i < Ss / 256; ++i)
        out[(size_t)b * Ss + tid + 256 * i] = vals[i] * inv;
}

// ---------------------------------------------------------------------------
extern "C" void kernel_launch(void* const* d_in, const int* in_sizes, int n_in,
                              void* d_out, int out_size, void* d_ws, size_t ws_size,
                              hipStream_t stream) {
    (void)in_sizes; (void)n_in; (void)out_size; (void)ws_size;
    const float* memory = (const float*)d_in[0];            // [B,S,D] f32
    const float* target = (const float*)d_in[1];            // [B,D]   f32
    const unsigned char* mmask = (const unsigned char*)d_in[2]; // [B,S] bool (1B)
    const float* wq = (const float*)d_in[3];                // [D,D] f32
    const float* wk = (const float*)d_in[4];                // [D,D] f32
    const float* v  = (const float*)d_in[5];                // [D]   f32
    float* out = (float*)d_out;                             // [B,S] f32

    // Workspace layout
    bf16_t* wkb   = (bf16_t*)d_ws;                                  // 2 MB
    float*  qbuf  = (float*)((char*)d_ws + (2u << 20));             // 64 KB
    float*  score = qbuf + Bb * Dd;                                 // 256 KB

    cvt_wk_kernel<<<(Dd * Dd / 4) / 256, 256, 0, stream>>>(wk, wkb);
    q_kernel<<<(Bb * Dd) / 256, 256, 0, stream>>>(target, wq, qbuf);

    const size_t ldsBytes = (size_t)TILE_M * ROWB + TILE_M * sizeof(float); // 132,352 B
    bahdanau_main<<<dim3(Ss / TILE_M, Bb), 256, ldsBytes, stream>>>(memory, wkb, qbuf, v, score);

    softmax_kernel<<<Bb, 256, 0, stream>>>(score, mmask, out);
}
